// SparseMoE_55559696941273
// MI455X (gfx1250) — matrix operand order repaired and checked by reading, not compiled, not run on hardware
//
#include <hip/hip_runtime.h>
#include <hip/hip_bf16.h>

typedef __attribute__((ext_vector_type(8)))  _Float16 v8h;
typedef __attribute__((ext_vector_type(16))) _Float16 v16h;
typedef __attribute__((ext_vector_type(8)))  float    v8f;

union H16 { v16h v; v8h h[2]; };

#define NTOK   16384
#define DDIM   512
#define HDIM   2048
#define NEXP   8
#define TOPK   2
#define TILE_M 32
#define HCHUNK 64

// ---------------- small init / convert kernels ----------------

__global__ void moe_zero_kernel(int* counts, float* accum) {
    int i = threadIdx.x;
    if (i < NEXP * TOPK) counts[i] = 0;
    if (i < 2 * NEXP + 1) accum[i] = 0.0f;
}

__global__ void moe_cvt_x_kernel(const float* __restrict__ x,
                                 _Float16* __restrict__ xh, int n) {
    int i = blockIdx.x * blockDim.x + threadIdx.x;
    if (i < n) xh[i] = (_Float16)x[i];
}

// W: (E, R, C) fp32  ->  Wt: (E, C, R) fp16   (Wt[e][c][r] = W[e][r][c])
__global__ void moe_transpose_w_kernel(const float* __restrict__ W,
                                       _Float16* __restrict__ Wt,
                                       int R, int C) {
    int i = blockIdx.x * blockDim.x + threadIdx.x;
    int rc_total = R * C;
    if (i >= NEXP * rc_total) return;
    int e  = i / rc_total;
    int rc = i - e * rc_total;
    int c  = rc / R;
    int r  = rc - c * R;
    Wt[i] = (_Float16)W[((size_t)e * R + r) * C + c];
}

// ---------------- router: logits, top-2, gates, dispatch lists ----------------

__global__ __launch_bounds__(256)
void moe_router_kernel(const float* __restrict__ x,
                       const float* __restrict__ Wg,
                       const float* __restrict__ bg,
                       int*   __restrict__ counts,
                       int*   __restrict__ lists,
                       float* __restrict__ gatel,
                       float* __restrict__ accum) {
    const int token = blockIdx.x * 8 + (threadIdx.x >> 5);  // one wave32 per token
    const int lane  = threadIdx.x & 31;

    float logit[NEXP];
#pragma unroll
    for (int e = 0; e < NEXP; ++e) logit[e] = 0.0f;

    const float* xr = x + (size_t)token * DDIM;
    for (int d = lane; d < DDIM; d += 32) {
        float xv = xr[d];
#pragma unroll
        for (int e = 0; e < NEXP; ++e) logit[e] += xv * Wg[d * NEXP + e];
    }
#pragma unroll
    for (int e = 0; e < NEXP; ++e)
        for (int off = 16; off > 0; off >>= 1)
            logit[e] += __shfl_down(logit[e], off, 32);

    if (lane == 0) {
#pragma unroll
        for (int e = 0; e < NEXP; ++e) logit[e] += bg[e];

        int i1 = 0;
#pragma unroll
        for (int e = 1; e < NEXP; ++e) if (logit[e] > logit[i1]) i1 = e;
        int i2 = (i1 == 0) ? 1 : 0;
#pragma unroll
        for (int e = 0; e < NEXP; ++e)
            if (e != i1 && logit[e] > logit[i2]) i2 = e;

        float v1 = logit[i1], v2 = logit[i2];
        float e2 = expf(v2 - v1);
        float g1 = 1.0f / (1.0f + e2);
        float g2 = e2 / (1.0f + e2);

        float ssum = 0.0f;
#pragma unroll
        for (int e = 0; e < NEXP; ++e) ssum += expf(logit[e] - v1);
        float lse = v1 + logf(ssum);

        atomicAdd(&accum[i1], g1);              // sum of gates per expert
        atomicAdd(&accum[i2], g2);
        atomicAdd(&accum[NEXP + i1], 1.0f);     // selection counts per expert
        atomicAdd(&accum[NEXP + i2], 1.0f);
        atomicAdd(&accum[2 * NEXP], lse * lse); // sum of logsumexp^2

        int p1 = atomicAdd(&counts[i1 * TOPK + 0], 1);
        lists[(i1 * TOPK + 0) * NTOK + p1] = token;
        gatel[(i1 * TOPK + 0) * NTOK + p1] = g1;
        int p2 = atomicAdd(&counts[i2 * TOPK + 1], 1);
        lists[(i2 * TOPK + 1) * NTOK + p2] = token;
        gatel[(i2 * TOPK + 1) * NTOK + p2] = g2;
    }
}

// ---------------- fused expert: relu(x@W1+b1)@W2+b2, gate-weighted ----------------

__global__ __launch_bounds__(256)
void moe_expert_kernel(const _Float16* __restrict__ xh,
                       const _Float16* __restrict__ w1t,  // (E,H,D)
                       const _Float16* __restrict__ w2t,  // (E,D,H)
                       const float*    __restrict__ b1,   // (E,H)
                       const float*    __restrict__ b2,   // (E,D)
                       const int*      __restrict__ lists,
                       const float*    __restrict__ gatel,
                       const int*      __restrict__ counts,
                       float*          __restrict__ out,
                       int kslot) {
    __shared__ _Float16 sA[TILE_M * DDIM];     // 32 KB staged x tile
    __shared__ _Float16 sHc[TILE_M * HCHUNK];  // 4 KB hidden chunk

    const int e        = blockIdx.y;
    const int tile     = blockIdx.x;
    const int listBase = (e * TOPK + kslot) * NTOK;
    const int cnt      = counts[e * TOPK + kslot];
    if (tile * TILE_M >= cnt) return;          // uniform over the block

    const int tid  = threadIdx.x;
    const int wave = tid >> 5;
    const int lane = tid & 31;
    const int ln   = lane & 15;   // fragment N-column / A-row index
    const int kg   = lane >> 4;   // fragment K-group (0/1)

    // ---- stage 32 gathered token rows of xh into LDS via async global->LDS DMA ----
    {
        const int row = tid >> 3;          // 0..31
        const int seg = tid & 7;           // 8 segments x 64 halfs
        int s   = tile * TILE_M + row;
        int tok = (s < cnt) ? lists[listBase + s] : lists[listBase];
        const _Float16* gsrc = xh + (size_t)tok * DDIM + seg * 64;
        unsigned ldsAddr =
            (unsigned)(unsigned long)(__attribute__((address_space(3))) void*)
                (sA + row * DDIM + seg * 64);
        // 8 x b128: contiguous on both sides; IOFFSET applies to LDS and memory
#pragma unroll
        for (int i = 0; i < 8; ++i) {
            asm volatile("global_load_async_to_lds_b128 %0, %1, off"
                         :: "v"(ldsAddr + i * 16), "v"(gsrc + i * 8)
                         : "memory");
        }
        asm volatile("s_wait_asynccnt 0x0" ::: "memory");
    }
    __syncthreads();

    const int mhalf = wave >> 2;   // which 16-row half of the tile
    const int quad  = wave & 3;    // GEMM1: 16-wide h slice; GEMM2: 128-wide d slice

    // tokens/gates for the rows this lane will store (C/D layout: M = v + 8*kg)
    int   tokv[8];
    float gatev[8];
#pragma unroll
    for (int v = 0; v < 8; ++v) {
        int s = tile * TILE_M + mhalf * 16 + v + 8 * kg;
        if (s < cnt) { tokv[v] = lists[listBase + s]; gatev[v] = gatel[listBase + s]; }
        else         { tokv[v] = -1;                  gatev[v] = 0.0f; }
    }

    const v8f vzero = {};
    v8f acc[8];                    // 32x512 y tile: this wave owns 16x128 -> 8 frags
#pragma unroll
    for (int f = 0; f < 8; ++f) acc[f] = vzero;

    const _Float16* w1e = w1t + (size_t)e * HDIM * DDIM;
    const _Float16* w2e = w2t + (size_t)e * DDIM * HDIM;
    const float*    b1e = b1 + e * HDIM;
    const float*    b2e = b2 + e * DDIM;

    for (int hc = 0; hc < HDIM / HCHUNK; ++hc) {
        const int hbase = hc * HCHUNK;

        // ---- GEMM1: 16x16 frag of h at rows mhalf*16.., cols hbase+quad*16.. ----
        v8f hacc = vzero;
        const int hcol = hbase + quad * 16 + ln;
        const _Float16* bcol = w1e + (size_t)hcol * DDIM;   // contiguous in d
        __builtin_prefetch(w1e + (size_t)(hbase + HCHUNK + quad * 16 + ln) * DDIM, 0, 0);
#pragma unroll 4
        for (int kk = 0; kk < DDIM; kk += 32) {
            H16 aU;
            const _Float16* arow = sA + (mhalf * 16 + ln) * DDIM + kk;
            aU.h[0] = *(const v8h*)(arow + kg * 16);         // K = kk + kg*16 .. +7
            aU.h[1] = *(const v8h*)(arow + kg * 16 + 8);     // K = kk+kg*16+8 .. +15
            v16h bf = *(const v16h*)(bcol + kk + kg * 16);   // K = kk + kg*16 .. +15
            hacc = __builtin_amdgcn_wmma_f32_16x16x32_f16(
                false, aU.v, false, bf, (short)0, hacc, false, false);
        }

        float b1v = b1e[hcol];
        __syncthreads();   // previous chunk's GEMM2 readers are done with sHc
#pragma unroll
        for (int v = 0; v < 8; ++v) {
            float hv = fmaxf(hacc[v] + b1v, 0.0f);           // bias + ReLU
            sHc[(mhalf * 16 + v + 8 * kg) * HCHUNK + quad * 16 + ln] = (_Float16)hv;
        }
        __syncthreads();

        // ---- GEMM2: accumulate y += h_chunk @ W2 over this 64-wide K slice ----
#pragma unroll
        for (int kk2 = 0; kk2 < HCHUNK; kk2 += 32) {
            H16 aU;
            const _Float16* arow = sHc + (mhalf * 16 + ln) * HCHUNK + kk2;
            aU.h[0] = *(const v8h*)(arow + kg * 16);
            aU.h[1] = *(const v8h*)(arow + kg * 16 + 8);
#pragma unroll
            for (int f = 0; f < 8; ++f) {
                int dcol = quad * 128 + f * 16 + ln;
                v16h bf = *(const v16h*)(w2e + (size_t)dcol * HDIM + hbase + kk2 + kg * 16);
                acc[f] = __builtin_amdgcn_wmma_f32_16x16x32_f16(
                    false, aU.v, false, bf, (short)0, acc[f], false, false);
            }
        }
    }

    // ---- epilogue: + b2, * gate, scatter to out (k=0 writes, k=1 adds) ----
#pragma unroll
    for (int f = 0; f < 8; ++f) {
        int   dcol = quad * 128 + f * 16 + ln;
        float b2v  = b2e[dcol];
#pragma unroll
        for (int v = 0; v < 8; ++v) {
            if (tokv[v] >= 0) {
                float  val = (acc[f][v] + b2v) * gatev[v];
                float* po  = out + (size_t)tokv[v] * DDIM + dcol;
                if (kslot == 0) *po = val;
                else            *po += val;
            }
        }
    }
}

// ---------------- losses ----------------

__global__ void moe_finalize_kernel(const float* __restrict__ accum,
                                    float* __restrict__ out_tail) {
    if (threadIdx.x == 0) {
        const float invN = 1.0f / (float)NTOK;
        float aux = 0.0f;
        for (int e = 0; e < NEXP; ++e)
            aux += (accum[e] * invN) * (accum[NEXP + e] * invN);
        out_tail[0] = (float)NEXP * aux;          // aux_loss
        out_tail[1] = accum[2 * NEXP] * invN;     // z_loss
    }
}

// ---------------- host-side launch ----------------

extern "C" void kernel_launch(void* const* d_in, const int* in_sizes, int n_in,
                              void* d_out, int out_size, void* d_ws, size_t ws_size,
                              hipStream_t stream) {
    const float* x  = (const float*)d_in[0];
    const float* Wg = (const float*)d_in[1];
    const float* bg = (const float*)d_in[2];
    const float* W1 = (const float*)d_in[3];
    const float* b1 = (const float*)d_in[4];
    const float* W2 = (const float*)d_in[5];
    const float* b2 = (const float*)d_in[6];
    float* out = (float*)d_out;

    char* ws = (char*)d_ws;
    size_t off = 0;
    _Float16* xh  = (_Float16*)(ws + off); off += (size_t)NTOK * DDIM * 2;
    _Float16* w1t = (_Float16*)(ws + off); off += (size_t)NEXP * HDIM * DDIM * 2;
    _Float16* w2t = (_Float16*)(ws + off); off += (size_t)NEXP * DDIM * HDIM * 2;
    int*   lists  = (int*)  (ws + off);    off += (size_t)NEXP * TOPK * NTOK * 4;
    float* gatel  = (float*)(ws + off);    off += (size_t)NEXP * TOPK * NTOK * 4;
    int*   counts = (int*)  (ws + off);    off += 256;
    float* accum  = (float*)(ws + off);    off += 256;

    moe_zero_kernel<<<1, 64, 0, stream>>>(counts, accum);

    {
        int n = NTOK * DDIM;
        moe_cvt_x_kernel<<<(n + 255) / 256, 256, 0, stream>>>(x, xh, n);
    }
    {
        int n = NEXP * DDIM * HDIM;
        moe_transpose_w_kernel<<<(n + 255) / 256, 256, 0, stream>>>(W1, w1t, DDIM, HDIM);
        moe_transpose_w_kernel<<<(n + 255) / 256, 256, 0, stream>>>(W2, w2t, HDIM, DDIM);
    }

    moe_router_kernel<<<NTOK / 8, 256, 0, stream>>>(x, Wg, bg, counts, lists, gatel, accum);

    dim3 egrid(NTOK / TILE_M, NEXP);
    moe_expert_kernel<<<egrid, 256, 0, stream>>>(xh, w1t, w2t, b1, b2,
                                                 lists, gatel, counts, out, 0);
    moe_expert_kernel<<<egrid, 256, 0, stream>>>(xh, w1t, w2t, b1, b2,
                                                 lists, gatel, counts, out, 1);

    moe_finalize_kernel<<<1, 32, 0, stream>>>(accum, out + (size_t)NTOK * DDIM);
}
